// ShiftReduceDP_68728066671296
// MI455X (gfx1250) — compile-verified, matrix operand order checked by reading, not run on hardware
//
#include <hip/hip_runtime.h>

#define S_DIM 64
#define B_DIM 8
#define H_DIM 512
#define V_DIM 8192
#define NEGV (-1.0e9f)
#define N_PAIR 1953           // i<j, i,j in 0..62
#define R_PAIR (N_PAIR * 8)   // 15624
#define R_TOT  (R_PAIR + 8)   // 15632 = 977 * 16 exactly
#define N_TILES_M 977

typedef __attribute__((ext_vector_type(16))) _Float16 v16h;
typedef __attribute__((ext_vector_type(8)))  _Float16 v8h;
typedef __attribute__((ext_vector_type(8)))  float    v8f;

// ---------------------------------------------------------------------------
// A/B fragment loader for V_WMMA_F32_16X16X32_F16 (wave32).
// 16-bit A 16x32 layout: lanes 0-15 hold K={k0+0..7, k0+16..23},
// lanes 16-31 hold K={k0+8..15, k0+24..31}; row/col = lane%16.
// ---------------------------------------------------------------------------
static __device__ inline v16h load_frag(const _Float16* base, int k0, int sel) {
  v8h lo = *reinterpret_cast<const v8h*>(base + k0 + sel);
  v8h hi = *reinterpret_cast<const v8h*>(base + k0 + 16 + sel);
  v16h r;
#pragma unroll
  for (int t = 0; t < 8; ++t) { r[t] = lo[t]; r[t + 8] = hi[t]; }
  return r;
}

// ---------------------------------------------------------------------------
// K0a: fp32 -> fp16 copy (row-major preserved)
// ---------------------------------------------------------------------------
__global__ void convert_f16_kernel(const float* __restrict__ src,
                                   _Float16* __restrict__ dst, int n) {
  int idx = blockIdx.x * 256 + threadIdx.x;
  if (idx < n) dst[idx] = (_Float16)src[idx];
}

// ---------------------------------------------------------------------------
// K0b: fp32 (K x N, row-major) -> fp16 (N x K, row-major)  [transpose+convert]
// so that one output column of the GEMM is a contiguous fp16 run of K values.
// ---------------------------------------------------------------------------
__global__ void transpose_f16_kernel(const float* __restrict__ src,
                                     _Float16* __restrict__ dst, int K, int N) {
  int idx = blockIdx.x * 256 + threadIdx.x;
  if (idx < K * N) {
    int k = idx / N;
    int n = idx - k * N;
    dst[n * K + k] = (_Float16)src[idx];
  }
}

// ---------------------------------------------------------------------------
// K1: 4 projection GEMMs (512x512 @ 512x512) via WMMA:
//   tA = h @ t_W1[:H], tB = h @ t_W1[H:], wP = h @ w_W1[:H], wQ = h @ w_W1[H:]
// One 16x16 tile per wave, K-loop of 16 x wmma_f32_16x16x32_f16.
// ---------------------------------------------------------------------------
__global__ __launch_bounds__(256) void proj_gemm_kernel(
    const _Float16* __restrict__ hf16, const _Float16* __restrict__ t1t,
    const _Float16* __restrict__ w1t, float* __restrict__ tA,
    float* __restrict__ tB, float* __restrict__ wP, float* __restrict__ wQ) {
  int tid = threadIdx.x;
  int lane = tid & 31, wv = tid >> 5;
  int id = blockIdx.x * 8 + wv;            // 0..4095 tiles
  int outsel = id >> 10;
  int rem = id & 1023;
  int tm = rem >> 5, tn = rem & 31;
  const _Float16* wt;
  float* out;
  switch (outsel) {
    case 0:  wt = t1t;               out = tA; break;
    case 1:  wt = t1t + 512 * 512;   out = tB; break;
    case 2:  wt = w1t;               out = wP; break;
    default: wt = w1t + 512 * 512;   out = wQ; break;
  }
  int l16 = lane & 15;
  int sel = (lane >> 4) << 3;              // 0 or 8
  const _Float16* arow = hf16 + (tm * 16 + l16) * 512;
  const _Float16* brow = wt + (tn * 16 + l16) * 512;
  v8f c = {0.f, 0.f, 0.f, 0.f, 0.f, 0.f, 0.f, 0.f};
#pragma unroll
  for (int kk = 0; kk < 16; ++kk) {
    int k0 = kk * 32;
    v16h a = load_frag(arow, k0, sel);
    v16h b = load_frag(brow, k0, sel);
    c = __builtin_amdgcn_wmma_f32_16x16x32_f16(false, a, false, b, (short)0, c,
                                               false, false);
  }
  // C layout: col = lane%16; row = v + 8*(lane/16)
  int col = tn * 16 + l16;
  int rbase = tm * 16 + ((lane >> 4) << 3);
#pragma unroll
  for (int v = 0; v < 8; ++v) out[(rbase + v) * 512 + col] = c[v];
}

// ---------------------------------------------------------------------------
// K2: t-scores for all (i,j,b): t = relu(tA_i + tB_j + b1) . t_W2 + b2
//     sh_lp = log_sigmoid(-t), re_lp = log_sigmoid(t); also wp := NEG.
// One wave per (i,j,b).
// ---------------------------------------------------------------------------
__global__ __launch_bounds__(256) void tscore_kernel(
    const float* __restrict__ tA, const float* __restrict__ tB,
    const float* __restrict__ t_b1, const float* __restrict__ t_W2,
    const float* __restrict__ t_b2, float* __restrict__ sh_lp,
    float* __restrict__ re_lp, float* __restrict__ wp) {
  int gid = blockIdx.x * 256 + threadIdx.x;
  int wid = gid >> 5, lane = gid & 31;     // wid in [0, 32768)
  int i = wid >> 9;
  int j = (wid >> 3) & 63;
  int b = wid & 7;
  const float* pa = tA + (i * 8 + b) * 512;
  const float* pb = tB + (j * 8 + b) * 512;
  float acc = 0.f;
  for (int c = lane; c < 512; c += 32) {
    float hv = fmaxf(pa[c] + pb[c] + t_b1[c], 0.f);
    acc += hv * t_W2[c];
  }
#pragma unroll
  for (int m = 16; m >= 1; m >>= 1) acc += __shfl_xor(acc, m, 32);
  if (lane == 0) {
    float t = acc + t_b2[0];
    float l1p = log1pf(__expf(-fabsf(t)));
    sh_lp[wid] = -(fmaxf(t, 0.f) + l1p);   // log_sigmoid(-t)
    re_lp[wid] = -(fmaxf(-t, 0.f) + l1p);  // log_sigmoid(t)
    wp[wid] = NEGV;
  }
}

// ---------------------------------------------------------------------------
// row decode: rows 0..15623 are (pair p, batch b); rows 15624..15631 are the
// init rows (i=j=0, batch b) used for table[0,1].
// ---------------------------------------------------------------------------
static __device__ inline void decode_row(int r, const int* __restrict__ sent,
                                         int& i, int& j, int& b, int& tok,
                                         int& isinit) {
  if (r >= R_PAIR) {
    i = 0; j = 0; b = r - R_PAIR; isinit = 1;
    tok = sent[1 * B_DIM + b];
    return;
  }
  isinit = 0;
  int p = r >> 3;
  b = r & 7;
  int a = 0, rem = p;
  while (rem >= 62 - a) { rem -= 62 - a; ++a; }
  i = a;
  j = a + 1 + rem;
  tok = sent[(j + 1) * B_DIM + b];
}

// ---------------------------------------------------------------------------
// K3: big word GEMM with streaming logsumexp + token gather.
// One workgroup (8 waves) per 16-row M-tile. Hidden rows (16x512 f16) are
// built once in LDS; each wave covers 1024 of V=8192 columns, processed as
// 16 groups of 4 n-tiles: inside the K-loop one A fragment (LDS) is reused
// by 4 WMMAs against 4 B fragments (global, constant-offset addressing),
// quadrupling the WMMA : ds-load ratio. Per-lane streaming (max,sum) + token
// logit, combined across the 16-lane halves that share each C-matrix row.
// ---------------------------------------------------------------------------
__global__ __launch_bounds__(256) void word_gemm_kernel(
    const float* __restrict__ wP, const float* __restrict__ wQ,
    const float* __restrict__ w_b1, const _Float16* __restrict__ w2t,
    const float* __restrict__ w_b2, const int* __restrict__ sent,
    const float* __restrict__ sh_lp, float* __restrict__ wp,
    float* __restrict__ initlp) {
  __shared__ _Float16 Ash[16 * 512];
  __shared__ int ri[16], rj[16], rb[16], rtok[16], rini[16];
  __shared__ float pm[8][16], ps[8][16], pt[8][16];

  int tid = threadIdx.x;
  int tile = blockIdx.x;
  if (tid < 16) {
    int i, j, b, tok, ini;
    decode_row(tile * 16 + tid, sent, i, j, b, tok, ini);
    ri[tid] = i; rj[tid] = j; rb[tid] = b; rtok[tid] = tok; rini[tid] = ini;
  }
  __syncthreads();
  for (int idx = tid; idx < 16 * 512; idx += 256) {
    int t = idx >> 9, c = idx & 511;
    int i = ri[t], j = rj[t], b = rb[t];
    float hv = wP[(i * 8 + b) * 512 + c] + wQ[(j * 8 + b) * 512 + c] + w_b1[c];
    Ash[idx] = (_Float16)fmaxf(hv, 0.f);
  }
  __syncthreads();

  int lane = tid & 31, wv = tid >> 5;
  int l16 = lane & 15;
  int sel = (lane >> 4) << 3;
  int rowoff = (lane >> 4) << 3;           // C rows: v + rowoff
  const _Float16* arow = Ash + l16 * 512;

  float m[8], s[8], tk[8];
#pragma unroll
  for (int v = 0; v < 8; ++v) { m[v] = -1.0e30f; s[v] = 0.f; tk[v] = 0.f; }

  for (int ng = 0; ng < 16; ++ng) {        // 16 groups of 4 n-tiles
    int nbase = wv * 1024 + ng * 64;
    const _Float16* bcol = w2t + (size_t)(nbase + l16) * 512;
    v8f acc[4];
#pragma unroll
    for (int c4 = 0; c4 < 4; ++c4)
      acc[c4] = (v8f){0.f, 0.f, 0.f, 0.f, 0.f, 0.f, 0.f, 0.f};
#pragma unroll
    for (int kk = 0; kk < 16; ++kk) {
      int k0 = kk * 32;
      v16h a = load_frag(arow, k0, sel);   // one LDS A-frag, reused 4x
      v16h b0 = load_frag(bcol, k0, sel);
      v16h b1 = load_frag(bcol + 16 * 512, k0, sel);
      v16h b2 = load_frag(bcol + 32 * 512, k0, sel);
      v16h b3 = load_frag(bcol + 48 * 512, k0, sel);
      acc[0] = __builtin_amdgcn_wmma_f32_16x16x32_f16(false, a, false, b0,
                                                      (short)0, acc[0], false,
                                                      false);
      acc[1] = __builtin_amdgcn_wmma_f32_16x16x32_f16(false, a, false, b1,
                                                      (short)0, acc[1], false,
                                                      false);
      acc[2] = __builtin_amdgcn_wmma_f32_16x16x32_f16(false, a, false, b2,
                                                      (short)0, acc[2], false,
                                                      false);
      acc[3] = __builtin_amdgcn_wmma_f32_16x16x32_f16(false, a, false, b3,
                                                      (short)0, acc[3], false,
                                                      false);
    }
#pragma unroll
    for (int c4 = 0; c4 < 4; ++c4) {
      int n = nbase + c4 * 16 + l16;       // this lane's output column
      float bias = w_b2[n];
#pragma unroll
      for (int v = 0; v < 8; ++v) {
        float x = acc[c4][v] + bias;
        tk[v] += (n == rtok[rowoff + v]) ? x : 0.f;
        float mn = fmaxf(m[v], x);
        s[v] = s[v] * __expf(m[v] - mn) + __expf(x - mn);
        m[v] = mn;
      }
    }
  }
  // combine across the 16 lanes that share each row (xor masks stay in half)
#pragma unroll
  for (int v = 0; v < 8; ++v) {
    for (int msk = 1; msk < 16; msk <<= 1) {
      float om = __shfl_xor(m[v], msk, 32);
      float os = __shfl_xor(s[v], msk, 32);
      float ot = __shfl_xor(tk[v], msk, 32);
      float mn = fmaxf(m[v], om);
      s[v] = s[v] * __expf(m[v] - mn) + os * __expf(om - mn);
      m[v] = mn;
      tk[v] += ot;
    }
  }
  if (l16 == 0) {
#pragma unroll
    for (int v = 0; v < 8; ++v) {
      int row = rowoff + v;
      pm[wv][row] = m[v]; ps[wv][row] = s[v]; pt[wv][row] = tk[v];
    }
  }
  __syncthreads();
  if (tid < 16) {
    float M = -1.0e30f, Ssum = 0.f, T = 0.f;
    for (int w = 0; w < 8; ++w) {
      float om = pm[w][tid], os = ps[w][tid];
      float mn = fmaxf(M, om);
      Ssum = Ssum * __expf(M - mn) + os * __expf(om - mn);
      M = mn;
      T += pt[w][tid];
    }
    float lse = M + __logf(Ssum);
    float wlp = T - lse;                   // tok_logit - lse
    if (rini[tid]) {
      initlp[rb[tid]] = wlp;
    } else {
      int idx = (ri[tid] * 64 + rj[tid]) * 8 + rb[tid];
      wp[idx] = sh_lp[idx] + wlp;
    }
  }
}

// ---------------------------------------------------------------------------
// K4: inside-algorithm DP, single persistent workgroup. Wave per (i,b), lane
// per split point m (replicating the reference's NEG-masked 63-way reduce).
// ---------------------------------------------------------------------------
__global__ __launch_bounds__(1024) void dp_kernel(
    const float* __restrict__ wp, const float* __restrict__ re_lp,
    const float* __restrict__ initlp, float* __restrict__ tab,
    float* __restrict__ out) {
  int tid = threadIdx.x;
  for (int idx = tid; idx < 64 * 64 * 8; idx += 1024) tab[idx] = NEGV;
  __threadfence_block();
  __syncthreads();
  if (tid < 8) tab[(0 * 64 + 1) * 8 + tid] = initlp[tid];
  for (int idx = tid; idx < 62 * 8; idx += 1024) {
    int jd = (idx >> 3) + 1, b = idx & 7;
    tab[(jd * 64 + jd + 1) * 8 + b] = 0.f;
  }
  __threadfence_block();
  __syncthreads();

  int lane = tid & 31, wv = tid >> 5;
  for (int g = 2; g <= 63; ++g) {
    int npairs = (64 - g) * 8;
    for (int pid = wv; pid < npairs; pid += 32) {
      int i = pid >> 3, b = pid & 7, j = i + g;
      int ma = lane + 1;                   // m in 1..32
      int mb = lane + 33;                  // m in 33..64 (64 excluded)
      float sa, sb;
      if (ma < g) {
        int k = i + ma;
        sa = tab[(i * 64 + k) * 8 + b] + tab[(k * 64 + j) * 8 + b] +
             wp[(i * 64 + k) * 8 + b] + re_lp[(k * 64 + j) * 8 + b];
      } else sa = NEGV;
      if (mb < g) {
        int k = i + mb;
        sb = tab[(i * 64 + k) * 8 + b] + tab[(k * 64 + j) * 8 + b] +
             wp[(i * 64 + k) * 8 + b] + re_lp[(k * 64 + j) * 8 + b];
      } else sb = (mb <= 63) ? NEGV : -3.0e38f;
      float mx = fmaxf(sa, sb);
#pragma unroll
      for (int msk = 16; msk >= 1; msk >>= 1)
        mx = fmaxf(mx, __shfl_xor(mx, msk, 32));
      float sum = __expf(sa - mx) + __expf(sb - mx);
#pragma unroll
      for (int msk = 16; msk >= 1; msk >>= 1) sum += __shfl_xor(sum, msk, 32);
      if (lane == 0) tab[(i * 64 + j) * 8 + b] = __logf(sum) + mx;
    }
    __threadfence_block();
    __syncthreads();
  }
  if (tid == 0) {
    float acc = 0.f;
    for (int b = 0; b < 8; ++b)
      acc += tab[(0 * 64 + 63) * 8 + b] + re_lp[(0 * 64 + 63) * 8 + b];
    out[0] = -acc;
  }
}

// ---------------------------------------------------------------------------
extern "C" void kernel_launch(void* const* d_in, const int* in_sizes, int n_in,
                              void* d_out, int out_size, void* d_ws,
                              size_t ws_size, hipStream_t stream) {
  const float* h     = (const float*)d_in[0];
  const int*   sent  = (const int*)d_in[1];
  const float* t_W1  = (const float*)d_in[2];
  const float* t_b1  = (const float*)d_in[3];
  const float* t_W2  = (const float*)d_in[4];
  const float* t_b2  = (const float*)d_in[5];
  const float* w_W1  = (const float*)d_in[6];
  const float* w_b1  = (const float*)d_in[7];
  const float* w_W2  = (const float*)d_in[8];
  const float* w_b2  = (const float*)d_in[9];
  float* out = (float*)d_out;

  char* ws = (char*)d_ws;
  size_t o = 0;
  auto carve = [&](size_t bytes) {
    size_t r = o;
    o += (bytes + 255) & ~(size_t)255;
    return r;
  };
  _Float16* hf16  = (_Float16*)(ws + carve(512 * 512 * 2));
  _Float16* t1t   = (_Float16*)(ws + carve(2 * 512 * 512 * 2));
  _Float16* w1t   = (_Float16*)(ws + carve(2 * 512 * 512 * 2));
  _Float16* w2t   = (_Float16*)(ws + carve((size_t)V_DIM * 512 * 2));
  float* tA       = (float*)(ws + carve(512 * 512 * 4));
  float* tB       = (float*)(ws + carve(512 * 512 * 4));
  float* wPp      = (float*)(ws + carve(512 * 512 * 4));
  float* wQp      = (float*)(ws + carve(512 * 512 * 4));
  float* sh_lp    = (float*)(ws + carve(64 * 64 * 8 * 4));
  float* re_lp    = (float*)(ws + carve(64 * 64 * 8 * 4));
  float* wp       = (float*)(ws + carve(64 * 64 * 8 * 4));
  float* initlp   = (float*)(ws + carve(8 * 4));
  float* tab      = (float*)(ws + carve(64 * 64 * 8 * 4));
  (void)ws_size; (void)in_sizes; (void)n_in; (void)out_size;

  // fp32 -> fp16 conversions / transposes
  convert_f16_kernel<<<(512 * 512 + 255) / 256, 256, 0, stream>>>(h, hf16,
                                                                  512 * 512);
  transpose_f16_kernel<<<(512 * 512 + 255) / 256, 256, 0, stream>>>(
      t_W1, t1t, 512, 512);
  transpose_f16_kernel<<<(512 * 512 + 255) / 256, 256, 0, stream>>>(
      t_W1 + 512 * 512, t1t + 512 * 512, 512, 512);
  transpose_f16_kernel<<<(512 * 512 + 255) / 256, 256, 0, stream>>>(
      w_W1, w1t, 512, 512);
  transpose_f16_kernel<<<(512 * 512 + 255) / 256, 256, 0, stream>>>(
      w_W1 + 512 * 512, w1t + 512 * 512, 512, 512);
  transpose_f16_kernel<<<(512 * V_DIM + 255) / 256, 256, 0, stream>>>(
      w_W2, w2t, 512, V_DIM);

  // projections (4 x 512^3 GEMM, WMMA)
  proj_gemm_kernel<<<512, 256, 0, stream>>>(hf16, t1t, w1t, tA, tB, wPp, wQp);

  // pair scorer -> sh_lp / re_lp, wp = NEG
  tscore_kernel<<<4096, 256, 0, stream>>>(tA, tB, t_b1, t_W2, t_b2, sh_lp,
                                          re_lp, wp);

  // big word GEMM (15632 x 512 x 8192, WMMA, 4-way n-blocked)
  word_gemm_kernel<<<N_TILES_M, 256, 0, stream>>>(wPp, wQp, w_b1, w2t, w_b2,
                                                  sent, sh_lp, wp, initlp);

  // inside DP + final scalar
  dp_kernel<<<1, 1024, 0, stream>>>(wp, re_lp, initlp, tab, out);
}
